// MultiHeadAttention_18580028523170
// MI455X (gfx1250) — matrix operand order repaired and checked by reading, not compiled, not run on hardware
//
#include <hip/hip_runtime.h>
#include <hip/hip_bf16.h>

// MHA forward, B=4 T=2048 C=1024 H=16 DH=64 — bf16 WMMA pipeline for gfx1250.
// v_wmma_f32_16x16x32_bf16 everywhere; GEMMs use a 2x-unrolled software
// pipeline (no register rotation); flash attention double-buffers K/V tiles
// in LDS via global_load_async_to_lds_b128 + s_wait_asynccnt.

typedef __bf16 bf16;
typedef __attribute__((ext_vector_type(16))) __bf16 v16bf;
typedef __attribute__((ext_vector_type(8)))  __bf16 v8bf;
typedef __attribute__((ext_vector_type(8)))  float  v8f;
typedef __attribute__((ext_vector_type(4)))  int    v4i;

constexpr int Bb  = 4;
constexpr int TT  = 2048;
constexpr int CC  = 1024;
constexpr int HH  = 16;
constexpr int DHD = 64;
constexpr int BT  = Bb * TT;          // 8192

// ---- CDNA5 async global->LDS (guarded; sync-copy fallback) ------------------
#if defined(__has_builtin)
#if __has_builtin(__builtin_amdgcn_global_load_async_to_lds_b128)
#define HAVE_ASYNC 1
#endif
#endif
#ifndef HAVE_ASYNC
#define HAVE_ASYNC 0
#endif

#if HAVE_ASYNC
// signature (probe-verified): (AS1 v4i*, AS3 v4i*, imm off, imm cpol)
typedef __attribute__((address_space(1))) v4i gv4i_t;
typedef __attribute__((address_space(3))) v4i lv4i_t;
// numeric casts: global generic addr == AS1 addr; LDS offset == low 32 bits
#define ASYNC_B128(g, l)                                                      \
  __builtin_amdgcn_global_load_async_to_lds_b128(                             \
      (gv4i_t*)(size_t)(g), (lv4i_t*)(unsigned)(size_t)(l), 0, 0)
#if __has_builtin(__builtin_amdgcn_s_wait_asynccnt)
#define WAIT_ASYNC(n) __builtin_amdgcn_s_wait_asynccnt(n)
#else
#define WAIT_ASYNC(n) asm volatile("s_wait_asynccnt %0" ::"i"(n) : "memory")
#endif
#else
#define WAIT_ASYNC(n) ((void)0)
#endif

union ABu { v16bf v; v8bf h[2]; };

// ---- fragment loaders (contiguous k = 16*(L>>4)+s, same for A and B) ---------------------------------
// A (16x32 bf16) row-major, ld elements. Lane L: row=L&15;
// elems 0..15 = K[kg..kg+15], kg=(L>>4)*16.
__device__ __forceinline__ v16bf load_a(const bf16* p0, int ld, int lane) {
  const int row = lane & 15;
  const int kg  = (lane >> 4) << 4;
  const bf16* p = p0 + (size_t)row * ld + kg;
  ABu f;
  f.h[0] = *(const v8bf*)(p);
  f.h[1] = *(const v8bf*)(p + 8);
  return f.v;
}
// B: lane holds 16 contiguous bf16 (16 consecutive K of one column).
__device__ __forceinline__ v16bf load_b16(const bf16* p) {
  ABu f;
  f.h[0] = *(const v8bf*)(p);
  f.h[1] = *(const v8bf*)(p + 8);
  return f.v;
}
__device__ __forceinline__ v8f wmma_bf16(v16bf a, v16bf b, v8f c) {
  return __builtin_amdgcn_wmma_f32_16x16x32_bf16(false, a, false, b,
                                                 (short)0, c, false, false);
}

// load one k-step (32 wide) of fragments: 2 A frags + 4 B frags.
// wb = B-layout base (already n-offset), nstride = B-layout row length.
__device__ __forceinline__ void load_step(const bf16* ar0, const bf16* ar1,
                                          const bf16* wb, int nstride, int k,
                                          int lane, int grp, int coll,
                                          v16bf& a0, v16bf& a1, v16bf* bfr) {
  a0 = load_a(ar0 + k, CC, lane);
  a1 = load_a(ar1 + k, CC, lane);
  const int kbv = (k >> 4) + grp;
#pragma unroll
  for (int nt = 0; nt < 4; ++nt)
    bfr[nt] = load_b16(wb + ((size_t)kbv * nstride + nt * 16 + coll) * 16);
}

#define WMMA8(A0, A1, BB)                                                     \
  _Pragma("unroll") for (int nt = 0; nt < 4; ++nt) {                          \
    acc[0][nt] = wmma_bf16(A0, BB[nt], acc[0][nt]);                           \
    acc[1][nt] = wmma_bf16(A1, BB[nt], acc[1][nt]);                           \
  }

// ---- conversion kernels -----------------------------------------------------
__global__ __launch_bounds__(256) void cvt_bf16(const float* __restrict__ in,
                                                bf16* __restrict__ out, int n) {
  int i = blockIdx.x * 256 + threadIdx.x;
  if (i < n) out[i] = (bf16)in[i];
}

// W (H,C,DH) -> Wb[h][kb][n][ki] = W[h][kb*16+ki][n]
__global__ __launch_bounds__(256) void cvt_w_qkv(const float* __restrict__ wq,
                                                 const float* __restrict__ wk,
                                                 const float* __restrict__ wv,
                                                 bf16* __restrict__ oq,
                                                 bf16* __restrict__ ok,
                                                 bf16* __restrict__ ov) {
  int i = blockIdx.x * 256 + threadIdx.x;
  const float* src = (blockIdx.y == 0) ? wq : (blockIdx.y == 1) ? wk : wv;
  bf16*        dst = (blockIdx.y == 0) ? oq : (blockIdx.y == 1) ? ok : ov;
  int ki = i & 15;
  int n  = (i >> 4)  & 63;
  int kb = (i >> 10) & 63;
  int h  = i >> 16;
  dst[i] = (bf16)src[((size_t)h * CC + kb * 16 + ki) * DHD + n];
}

// Wo (C,C) -> Wob[kb][n][ki] = Wo[kb*16+ki][n]
__global__ __launch_bounds__(256) void cvt_wo(const float* __restrict__ wo,
                                              bf16* __restrict__ dst) {
  int i = blockIdx.x * 256 + threadIdx.x;
  int ki = i & 15;
  int n  = (i >> 4) & 1023;
  int kb = i >> 14;
  dst[i] = (bf16)wo[(size_t)(kb * 16 + ki) * CC + n];
}

// ---- QKV projection: 2x-unrolled pipeline, wave tile 32x64 -----------------
// grid = (BT/256, H, 3); 8 waves/WG; wave: rows [m0,m0+32) x all 64 d-cols.
__global__ __launch_bounds__(256) void qkv_gemm(const bf16* __restrict__ xb,
                                                const bf16* __restrict__ wqb,
                                                const bf16* __restrict__ wkb,
                                                const bf16* __restrict__ wvb,
                                                bf16* __restrict__ qb,
                                                bf16* __restrict__ kb,
                                                bf16* __restrict__ vtb) {
  const int lane = threadIdx.x & 31;
  const int wave = threadIdx.x >> 5;
  const int m0   = blockIdx.x * 256 + wave * 32;
  const int h    = blockIdx.y;
  const int which = blockIdx.z;
  const bf16* wb = (which == 0) ? wqb : (which == 1) ? wkb : wvb;
  wb += (size_t)h * CC * DHD;
  const int grp = lane >> 4, coll = lane & 15, rbase = grp << 3;
  const bf16* ar0 = xb + (size_t)m0 * CC;
  const bf16* ar1 = xb + (size_t)(m0 + 16) * CC;

  v8f acc[2][4] = {};
  v16bf a0A, a1A, bA[4], a0B, a1B, bB[4];
  load_step(ar0, ar1, wb, DHD, 0, lane, grp, coll, a0A, a1A, bA);
  // two k-steps (64 wide) per iteration; register sets alternate, no copies
  for (int k0 = 0; k0 < CC - 64; k0 += 64) {
    load_step(ar0, ar1, wb, DHD, k0 + 32, lane, grp, coll, a0B, a1B, bB);
    WMMA8(a0A, a1A, bA);                       // step k0
    load_step(ar0, ar1, wb, DHD, k0 + 64, lane, grp, coll, a0A, a1A, bA);
    WMMA8(a0B, a1B, bB);                       // step k0+32
  }
  load_step(ar0, ar1, wb, DHD, CC - 32, lane, grp, coll, a0B, a1B, bB);
  WMMA8(a0A, a1A, bA);                         // step CC-64
  WMMA8(a0B, a1B, bB);                         // step CC-32

  const int bidx = m0 / TT;                    // 256-row tile never crosses T
  const size_t bh = (size_t)bidx * HH + h;
#pragma unroll
  for (int rg = 0; rg < 2; ++rg) {
    const int gm = m0 + rg * 16 + rbase;
#pragma unroll
    for (int r = 0; r < 8; ++r) {
      const int t = (gm + r) % TT;
#pragma unroll
      for (int nt = 0; nt < 4; ++nt) {
        const int d = nt * 16 + coll;
        const bf16 val = (bf16)acc[rg][nt][r];
        if (which == 0)      qb [(bh * TT + t) * DHD + d] = val;
        else if (which == 1) kb [(bh * TT + t) * DHD + d] = val;
        else                 vtb[(bh * DHD + d) * TT + t] = val;  // V transposed
      }
    }
  }
}

// ---- stage one 64-key K tile (64x64) + Vt tile (64x64) into LDS -------------
__device__ __forceinline__ void stage_kv(const bf16* __restrict__ K,
                                         const bf16* __restrict__ Vt, int s0,
                                         bf16* kd, bf16* vd, int tid) {
  const bf16* ks = K + (size_t)s0 * DHD;     // 8 KB linear
#pragma unroll
  for (int c = tid; c < 512; c += 256) {     // 512 x 16B chunks each
    const int row = c >> 3;                  // 0..63
    const int col = (c & 7) << 3;            // 0..56 elements
#if HAVE_ASYNC
    ASYNC_B128(ks + c * 8,                       kd + c * 8);
    ASYNC_B128(Vt + (size_t)row * TT + s0 + col, vd + row * 64 + col);
#else
    *(v8bf*)(kd + c * 8)          = *(const v8bf*)(ks + c * 8);
    *(v8bf*)(vd + row * 64 + col) = *(const v8bf*)(Vt + (size_t)row * TT + s0 + col);
#endif
  }
}

// ---- flash attention, block-cooperative LDS staging -------------------------
// grid = (T/128, H, B); 8 waves; wave owns 16 query rows; 64 keys / iter;
// uniform key loop per block (barrier-safe), data-level causal mask.
__global__ __launch_bounds__(256) void attn_fwd(const bf16* __restrict__ qb,
                                                const bf16* __restrict__ kbuf,
                                                const bf16* __restrict__ vtb,
                                                bf16* __restrict__ attnb) {
  __shared__ bf16 ldsK[2][64 * DHD];         // 2 x 8 KB
  __shared__ bf16 ldsV[2][DHD * 64];         // 2 x 8 KB
  __shared__ bf16 ldsP[8][16 * 64];          // 16 KB, wave-private P staging
  const int tid = threadIdx.x;
  const int lane = tid & 31, wave = tid >> 5;
  const int h = blockIdx.y, b = blockIdx.z;
  const int tblk = blockIdx.x * 128;
  const int t0 = tblk + wave * 16;
  const size_t bh = (size_t)b * HH + h;
  const bf16* Q  = qb   + bh * TT * DHD;
  const bf16* K  = kbuf + bh * TT * DHD;
  const bf16* Vt = vtb  + bh * DHD * TT;
  const int grp = lane >> 4, coll = lane & 15, rbase = grp << 3;

  const v16bf qa0 = load_a(Q + (size_t)t0 * DHD + 0,  DHD, lane);
  const v16bf qa1 = load_a(Q + (size_t)t0 * DHD + 32, DHD, lane);

  float m[8], l[8];
  v8f acc[4] = {};
#pragma unroll
  for (int r = 0; r < 8; ++r) { m[r] = -1.0e30f; l[r] = 0.0f; }

  const int n_iter = (tblk >> 6) + 2;        // keys [0, tblk+128)
  stage_kv(K, Vt, 0, ldsK[0], ldsV[0], tid);

  for (int it = 0; it < n_iter; ++it) {
    const int s0 = it << 6;
    bf16* kd = ldsK[it & 1];
    bf16* vd = ldsV[it & 1];
    if (it + 1 < n_iter) {                   // DMA next tile, overlap compute
      stage_kv(K, Vt, s0 + 64, ldsK[(it + 1) & 1], ldsV[(it + 1) & 1], tid);
      WAIT_ASYNC(4);                         // only current tile must be done
    } else {
      WAIT_ASYNC(0);
    }
    __syncthreads();                         // tiles visible to all waves

    // --- scores S = Q K^T: batch 8 ds-frag loads, then 8 WMMAs --------------
    v16bf kfr[8];
#pragma unroll
    for (int nt = 0; nt < 4; ++nt) {
      const bf16* kp = kd + (nt * 16 + coll) * DHD + (grp << 4);
      kfr[2 * nt + 0] = load_b16(kp);
      kfr[2 * nt + 1] = load_b16(kp + 32);
    }
    v8f sc[4] = {};
#pragma unroll
    for (int nt = 0; nt < 4; ++nt) {
      sc[nt] = wmma_bf16(qa0, kfr[2 * nt + 0], sc[nt]);
      sc[nt] = wmma_bf16(qa1, kfr[2 * nt + 1], sc[nt]);
    }
    // --- V fragments early: ds loads overlap softmax VALU -------------------
    v16bf vfr[8];
#pragma unroll
    for (int nt = 0; nt < 4; ++nt) {
      const bf16* vp = vd + (nt * 16 + coll) * 64 + (grp << 4);
      vfr[2 * nt + 0] = load_b16(vp);
      vfr[2 * nt + 1] = load_b16(vp + 32);
    }
    // --- scale + causal mask (data-level; EXEC all-ones) --------------------
    const float scale = 0.125f;              // DH^-0.5
#pragma unroll
    for (int nt = 0; nt < 4; ++nt) {
      const int s = s0 + nt * 16 + coll;
#pragma unroll
      for (int r = 0; r < 8; ++r) {
        const int row = t0 + rbase + r;
        const float v = sc[nt][r] * scale;
        sc[nt][r] = (s <= row) ? v : -3.0e30f;
      }
    }
    // --- online softmax -----------------------------------------------------
#pragma unroll
    for (int r = 0; r < 8; ++r) {
      float v = fmaxf(fmaxf(sc[0][r], sc[1][r]), fmaxf(sc[2][r], sc[3][r]));
#pragma unroll
      for (int off = 1; off < 16; off <<= 1) v = fmaxf(v, __shfl_xor(v, off, 32));
      const float mn = fmaxf(m[r], v);
      float sm = 0.0f;
#pragma unroll
      for (int nt = 0; nt < 4; ++nt) {
        const float p = __expf(sc[nt][r] - mn);
        sc[nt][r] = p;
        sm += p;
      }
#pragma unroll
      for (int off = 1; off < 16; off <<= 1) sm += __shfl_xor(sm, off, 32);
      const float corr = __expf(m[r] - mn);
      l[r] = l[r] * corr + sm;
      m[r] = mn;
#pragma unroll
      for (int nt = 0; nt < 4; ++nt) acc[nt][r] *= corr;
    }
    // --- P: C-layout -> A-layout via wave-private LDS tile ------------------
    bf16* pw = ldsP[wave];
#pragma unroll
    for (int nt = 0; nt < 4; ++nt)
#pragma unroll
      for (int r = 0; r < 8; ++r)
        pw[(rbase + r) * 64 + nt * 16 + coll] = (bf16)sc[nt][r];
    const v16bf pa0 = load_a(pw + 0,  64, lane);
    const v16bf pa1 = load_a(pw + 32, 64, lane);
    // --- O += P V -----------------------------------------------------------
#pragma unroll
    for (int nt = 0; nt < 4; ++nt) {
      acc[nt] = wmma_bf16(pa0, vfr[2 * nt + 0], acc[nt]);
      acc[nt] = wmma_bf16(pa1, vfr[2 * nt + 1], acc[nt]);
    }
    __syncthreads();                         // all done reading buf before DMA reuse
  }
  // normalize + store concat-head (B,T,C) bf16
#pragma unroll
  for (int r = 0; r < 8; ++r) {
    const float inv = 1.0f / l[r];
    const int t = t0 + rbase + r;
#pragma unroll
    for (int nt = 0; nt < 4; ++nt)
      attnb[((size_t)b * TT + t) * CC + h * DHD + nt * 16 + coll] =
          (bf16)(acc[nt][r] * inv);
  }
}

// ---- output projection: 2x-unrolled pipeline, wave tile 32x64 ---------------
// grid = (BT/256, C/64)
__global__ __launch_bounds__(256) void out_proj(const bf16* __restrict__ attnb,
                                                const bf16* __restrict__ wob,
                                                const float* __restrict__ bo,
                                                float* __restrict__ out) {
  const int lane = threadIdx.x & 31;
  const int wave = threadIdx.x >> 5;
  const int m0 = blockIdx.x * 256 + wave * 32;
  const int n0 = blockIdx.y * 64;
  const int grp = lane >> 4, coll = lane & 15, rbase = grp << 3;
  const bf16* ar0 = attnb + (size_t)m0 * CC;
  const bf16* ar1 = attnb + (size_t)(m0 + 16) * CC;
  const bf16* wbp = wob + (size_t)n0 * 16;   // n-offset folded into base

  v8f acc[2][4] = {};
  v16bf a0A, a1A, bA[4], a0B, a1B, bB[4];
  load_step(ar0, ar1, wbp, CC, 0, lane, grp, coll, a0A, a1A, bA);
  for (int k0 = 0; k0 < CC - 64; k0 += 64) {
    load_step(ar0, ar1, wbp, CC, k0 + 32, lane, grp, coll, a0B, a1B, bB);
    WMMA8(a0A, a1A, bA);
    load_step(ar0, ar1, wbp, CC, k0 + 64, lane, grp, coll, a0A, a1A, bA);
    WMMA8(a0B, a1B, bB);
  }
  load_step(ar0, ar1, wbp, CC, CC - 32, lane, grp, coll, a0B, a1B, bB);
  WMMA8(a0A, a1A, bA);
  WMMA8(a0B, a1B, bB);

#pragma unroll
  for (int rg = 0; rg < 2; ++rg)
#pragma unroll
    for (int nt = 0; nt < 4; ++nt) {
      const int n = n0 + nt * 16 + coll;
      const float bias = bo[n];
#pragma unroll
      for (int r = 0; r < 8; ++r)
        out[(size_t)(m0 + rg * 16 + rbase + r) * CC + n] = acc[rg][nt][r] + bias;
    }
}

// ---- host launcher ----------------------------------------------------------
extern "C" void kernel_launch(void* const* d_in, const int* in_sizes, int n_in,
                              void* d_out, int out_size, void* d_ws, size_t ws_size,
                              hipStream_t stream) {
  (void)in_sizes; (void)n_in; (void)out_size; (void)ws_size;
  const float* x  = (const float*)d_in[0];
  const float* Wq = (const float*)d_in[1];
  const float* Wk = (const float*)d_in[2];
  const float* Wv = (const float*)d_in[3];
  const float* Wo = (const float*)d_in[4];
  const float* bo = (const float*)d_in[5];

  size_t off = 0;
  auto carve = [&](size_t bytes) -> char* {
    char* p = (char*)d_ws + off;
    off += (bytes + 255) & ~(size_t)255;
    return p;
  };
  bf16* xb   = (bf16*)carve((size_t)BT * CC * 2);            // 16 MB
  bf16* wqb  = (bf16*)carve((size_t)HH * CC * DHD * 2);      //  2 MB
  bf16* wkb  = (bf16*)carve((size_t)HH * CC * DHD * 2);
  bf16* wvb  = (bf16*)carve((size_t)HH * CC * DHD * 2);
  bf16* wob  = (bf16*)carve((size_t)CC * CC * 2);            //  2 MB
  bf16* qbuf = (bf16*)carve((size_t)Bb * HH * TT * DHD * 2); // 16 MB
  bf16* kbuf = (bf16*)carve((size_t)Bb * HH * TT * DHD * 2);
  bf16* vtbf = (bf16*)carve((size_t)Bb * HH * TT * DHD * 2);
  bf16* attn = (bf16*)carve((size_t)BT * CC * 2);            // ~92 MB total

  cvt_bf16<<<(BT * CC) / 256, 256, 0, stream>>>(x, xb, BT * CC);
  cvt_w_qkv<<<dim3((HH * CC * DHD) / 256, 3), 256, 0, stream>>>(
      Wq, Wk, Wv, wqb, wkb, wvb);
  cvt_wo<<<(CC * CC) / 256, 256, 0, stream>>>(Wo, wob);

  qkv_gemm<<<dim3(BT / 256, HH, 3), 256, 0, stream>>>(
      xb, wqb, wkb, wvb, qbuf, kbuf, vtbf);

  attn_fwd<<<dim3(TT / 128, HH, Bb), 256, 0, stream>>>(qbuf, kbuf, vtbf, attn);

  out_proj<<<dim3(BT / 256, CC / 64), 256, 0, stream>>>(
      attn, wob, bo, (float*)d_out);
}
